// MultiTaskGNN_22101901705501
// MI455X (gfx1250) — matrix operand order repaired and checked by reading, not compiled, not run on hardware
//
#include <hip/hip_runtime.h>

#define HH 3  // attention heads

typedef __attribute__((ext_vector_type(16))) __bf16 v16bf;
typedef __attribute__((ext_vector_type(8)))  __bf16 v8bf;
typedef __attribute__((ext_vector_type(8)))  float  v8f;

// ---------- helpers: orderable float encoding for atomic max ----------
__device__ __forceinline__ unsigned f2ord(float f) {
  unsigned u = __float_as_uint(f);
  return (u & 0x80000000u) ? ~u : (u | 0x80000000u);
}
__device__ __forceinline__ float ord2f(unsigned u) {
  return (u & 0x80000000u) ? __uint_as_float(u & 0x7FFFFFFFu) : __uint_as_float(~u);
}

// ---------- tiny utility kernels ----------
__global__ void k_zero_f32(float* __restrict__ p, int n) {
  int i = blockIdx.x * blockDim.x + threadIdx.x;
  if (i < n) p[i] = 0.f;
}
__global__ void k_fill_u32(unsigned* __restrict__ p, unsigned v, int n) {
  int i = blockIdx.x * blockDim.x + threadIdx.x;
  if (i < n) p[i] = v;
}
__global__ void k_copy(const float* __restrict__ s, float* __restrict__ d, int n) {
  int i = blockIdx.x * blockDim.x + threadIdx.x;
  if (i < n) d[i] = s[i];
}
__global__ void k_copy_col(const float* __restrict__ x, float* __restrict__ w, int N, int D, int col) {
  int i = blockIdx.x * blockDim.x + threadIdx.x;
  if (i < N) w[i] = x[(size_t)i * D + col];
}

// ---------- pack fp32 weight W[K x Nc] into WMMA-B fragment order ----------
// Bp layout: [ct][ks][lane][16] bf16, ct = 16-col tile, ks = 32-wide K step.
// lane: hi = lane>>4 selects K half-group, col = ct*16 + (lane&15).
// j<8 : kb = ks*32 + hi*8 + j ; j>=8 : kb = ks*32 + hi*8 + 8 + j   (== +16 + hi*8 + (j-8))
__global__ void k_pack_B(const float* __restrict__ W, __bf16* __restrict__ Bp,
                         int K, int Nc, int KP) {
  int nct = (Nc + 15) >> 4, ksteps = KP >> 5;
  int total = nct * ksteps * 512;
  int i = blockIdx.x * blockDim.x + threadIdx.x;
  if (i >= total) return;
  int j    = i & 15;
  int lane = (i >> 4) & 31;
  int ks   = (i >> 9) % ksteps;
  int ct   = (i >> 9) / ksteps;
  int hi = lane >> 4, colr = lane & 15;
  int kb  = ks * 32 + hi * 8 + ((j < 8) ? j : (j + 8));
  int col = ct * 16 + colr;
  float v = (kb < K && col < Nc) ? W[(size_t)kb * Nc + col] : 0.f;
  Bp[i] = (__bf16)v;
}

// ---------- WMMA GEMM: C[M x ldc] = diag(rowScale) * A[M x K] @ B[K x Nc] (+bias, relu) ----
// block = 128 threads (4 waves). grid = (M/16, ceil(ceil(Nc/16)/4)).
// Preconditions: M % 16 == 0; ldc % 16 == 0; C is a padded scratch buffer of M x ldc;
// bias only used when Nc % 16 == 0. Stores are fully branchless.
template <int KP>
__global__ void gemm_bf16_wmma(const float* __restrict__ A, const float* __restrict__ rowScale,
                               const __bf16* __restrict__ Bp, const float* __restrict__ bias,
                               float* __restrict__ C, int M, int K, int ldc, int relu) {
  constexpr int KSTEPS = KP / 32;
  __shared__ __align__(16) __bf16 As[16 * KP];
  const int row0 = blockIdx.x * 16;

  // branchless cooperative A-tile stage: fp32 -> bf16 LDS, zero pad k >= K
  for (int idx = threadIdx.x; idx < 16 * KP; idx += blockDim.x) {
    int r = idx / KP, k = idx - r * KP;
    int gr = row0 + r;  // gr < M by precondition
    int kc = (k < K) ? k : 0;
    float v = A[(size_t)gr * K + kc];
    if (rowScale) v *= rowScale[gr];
    if (k >= K) v = 0.f;
    As[idx] = (__bf16)v;
  }
  __syncthreads();

  const int wave = threadIdx.x >> 5;
  const int lane = threadIdx.x & 31;
  const int ct   = blockIdx.y * 4 + wave;  // 16-column tile index
  if (ct * 16 >= ldc) return;
  const int r = lane & 15, hi = lane >> 4;

  v8f acc = {};
#pragma unroll
  for (int ks = 0; ks < KSTEPS; ++ks) {
    const __bf16* ap = &As[r * KP + ks * 32 + hi * 8];
    v8bf alo = *(const v8bf*)ap;         // K = ks*32 + hi*8 .. +8
    v8bf ahi = *(const v8bf*)(ap + 16);  // K = ks*32 + 16 + hi*8 .. +8
    v16bf a = __builtin_shufflevector(alo, ahi, 0, 1, 2, 3, 4, 5, 6, 7,
                                      8, 9, 10, 11, 12, 13, 14, 15);
    v16bf b = *(const v16bf*)(Bp + (((size_t)ct * KSTEPS + ks) * 32 + lane) * 16);
    acc = __builtin_amdgcn_wmma_f32_16x16x32_bf16(false, a, false, b, (short)0, acc, false, false);
  }

  const int gc = ct * 16 + r;
  const float badd = bias ? bias[gc] : 0.f;
  float* cp = C + (size_t)(row0 + hi * 8) * ldc + gc;
#pragma unroll
  for (int v = 0; v < 8; ++v) {
    float o = acc[v] + badd;
    if (relu) o = fmaxf(o, 0.f);
    cp[(size_t)v * ldc] = o;
  }
}

// ---------- GAT attention kernels ----------
__global__ void k_el_er(const float* __restrict__ z, const float* __restrict__ al, const float* __restrict__ ar,
                        float* __restrict__ el, float* __restrict__ er, int N, int Dh, int ldz) {
  int i = blockIdx.x * blockDim.x + threadIdx.x;
  if (i >= N * HH) return;
  int n = i / HH, h = i - n * HH;
  const float* zr  = z + (size_t)n * ldz + h * Dh;
  const float* alr = al + h * Dh;
  const float* arr = ar + h * Dh;
  float sl = 0.f, sr = 0.f;
  for (int d = 0; d < Dh; ++d) {
    float v = zr[d];
    sl += v * alr[d];
    sr += v * arr[d];
  }
  el[i] = sl;
  er[i] = sr;
}

__global__ void k_edge_max(const float* __restrict__ el, const float* __restrict__ er,
                           const int* __restrict__ src, const int* __restrict__ dst,
                           float* __restrict__ e, unsigned* __restrict__ m, int E) {
  int i = blockIdx.x * blockDim.x + threadIdx.x;
  if (i >= E * HH) return;
  int ed = i / HH, h = i - ed * HH;
  float v = el[src[ed] * HH + h] + er[dst[ed] * HH + h];
  v = (v > 0.f) ? v : 0.2f * v;  // leaky_relu(0.2)
  e[i] = v;
  atomicMax(&m[dst[ed] * HH + h], f2ord(v));
}

__global__ void k_edge_exp(float* __restrict__ e, const unsigned* __restrict__ m,
                           const int* __restrict__ dst, float* __restrict__ den, int E) {
  int i = blockIdx.x * blockDim.x + threadIdx.x;
  if (i >= E * HH) return;
  int ed = i / HH, h = i - ed * HH;
  float ee = __expf(e[i] - ord2f(m[dst[ed] * HH + h]));
  e[i] = ee;
  atomicAdd(&den[dst[ed] * HH + h], ee);
}

// one block per edge; 192 threads cover H*Dh features
__global__ void k_edge_aggr(const float* __restrict__ ee, const float* __restrict__ den,
                            const int* __restrict__ src, const int* __restrict__ dst,
                            const float* __restrict__ z, float* __restrict__ agg,
                            float* __restrict__ attn, int E, int Dh, int ldz) {
  int ed = blockIdx.x;
  int f  = threadIdx.x;
  if (ed + 1 < E) __builtin_prefetch(&src[ed + 1], 0, 1);
  int s = src[ed], d = dst[ed];
  int HD = HH * Dh;
  if (f < HD) {
    int h = f / Dh;
    float alpha = ee[ed * HH + h] / den[d * HH + h];
    atomicAdd(&agg[(size_t)d * HD + f], alpha * z[(size_t)s * ldz + f]);
  }
  if (f == 0 && attn) {
    float mx = -3.4e38f;
    for (int h = 0; h < HH; ++h) mx = fmaxf(mx, ee[ed * HH + h] / den[d * HH + h]);
    attn[ed] = mx;
  }
}

__global__ void k_bias_headmax(const float* __restrict__ agg, const float* __restrict__ b,
                               float* __restrict__ x, int N, int Dh, int relu) {
  int i = blockIdx.x * blockDim.x + threadIdx.x;
  if (i >= N * Dh) return;
  int n = i / Dh, d = i - n * Dh;
  int HD = HH * Dh;
  float mx = -3.4e38f;
  for (int h = 0; h < HH; ++h) mx = fmaxf(mx, agg[(size_t)n * HD + h * Dh + d] + b[h * Dh + d]);
  if (relu) mx = fmaxf(mx, 0.f);
  x[(size_t)n * Dh + d] = mx;
}

// ---------- GCN kernels ----------
__global__ void k_degrees(const int* __restrict__ src, const int* __restrict__ dst,
                          float* __restrict__ dout, float* __restrict__ din, int E) {
  int i = blockIdx.x * blockDim.x + threadIdx.x;
  if (i < E) {
    atomicAdd(&dout[src[i]], 1.f);
    atomicAdd(&din[dst[i]], 1.f);
  }
}
__global__ void k_rsqrt_clip(float* __restrict__ p, int n) {
  int i = blockIdx.x * blockDim.x + threadIdx.x;
  if (i < n) p[i] = rsqrtf(fmaxf(p[i], 1.f));
}
// one block per edge, D threads
__global__ void k_gcn_scatter(const float* __restrict__ x, const float* __restrict__ rsout,
                              const int* __restrict__ src, const int* __restrict__ dst,
                              float* __restrict__ agg, int E, int D) {
  int ed = blockIdx.x;
  int f  = threadIdx.x;
  if (ed + 1 < E) __builtin_prefetch(&src[ed + 1], 0, 1);
  if (f < D) {
    int s = src[ed], d = dst[ed];
    atomicAdd(&agg[(size_t)d * D + f], x[(size_t)s * D + f] * rsout[s]);
  }
}

// ---------- pooling / readout ----------
__global__ void k_pool_sum(const float* __restrict__ x, const int* __restrict__ gid,
                           float* __restrict__ sum, float* __restrict__ cnt, int N, int D) {
  int i = blockIdx.x * blockDim.x + threadIdx.x;
  if (i >= N * D) return;
  int n = i / D, d = i - n * D;
  int g = gid[n];
  atomicAdd(&sum[g * D + d], x[(size_t)n * D + d]);
  if (d == 0) atomicAdd(&cnt[g], 1.f);
}
__global__ void k_pool_div(float* __restrict__ sum, const float* __restrict__ cnt, int G_, int D) {
  int i = blockIdx.x * blockDim.x + threadIdx.x;
  if (i >= G_ * D) return;
  int g = i / D;
  sum[i] = sum[i] / fmaxf(cnt[g], 1.f);
}
__global__ void k_readout(const float* __restrict__ hg, const float* __restrict__ W,
                          const float* __restrict__ b, float* __restrict__ out, int G_, int D, int O) {
  int i = blockIdx.x * blockDim.x + threadIdx.x;
  if (i >= G_ * O) return;
  int g = i / O, o = i - g * O;
  float s = b[o];
  for (int d = 0; d < D; ++d) s += hg[g * D + d] * W[d * O + o];
  out[i] = s;
}

// ---------------------------------------------------------------------------
extern "C" void kernel_launch(void* const* d_in, const int* in_sizes, int n_in,
                              void* d_out_v, int out_size, void* d_ws, size_t ws_size,
                              hipStream_t stream) {
  (void)in_sizes; (void)n_in; (void)out_size; (void)ws_size;
  const int N = 40000, E = 400000, G = 100;

  const float* h1 = (const float*)d_in[0];
  const float* h2 = (const float*)d_in[1];
  const float* h3 = (const float*)d_in[2];
  const int* src1 = (const int*)d_in[3];  const int* dst1 = (const int*)d_in[4];  const int* gid1 = (const int*)d_in[5];
  const int* src2 = (const int*)d_in[6];  const int* dst2 = (const int*)d_in[7];  const int* gid2 = (const int*)d_in[8];
  const int* src3 = (const int*)d_in[9];  const int* dst3 = (const int*)d_in[10]; const int* gid3 = (const int*)d_in[11];
  const float* Wg1 = (const float*)d_in[12]; const float* al1 = (const float*)d_in[13];
  const float* ar1 = (const float*)d_in[14]; const float* bg1 = (const float*)d_in[15];
  const float* Wg2 = (const float*)d_in[16]; const float* al2 = (const float*)d_in[17];
  const float* ar2 = (const float*)d_in[18]; const float* bg2 = (const float*)d_in[19];
  const float* Wc1 = (const float*)d_in[20]; const float* bc1 = (const float*)d_in[21];
  const float* Wc2 = (const float*)d_in[22]; const float* bc2 = (const float*)d_in[23];
  const float* Wc3 = (const float*)d_in[24]; const float* bc3 = (const float*)d_in[25];
  const float* Wr1 = (const float*)d_in[26]; const float* br1 = (const float*)d_in[27];
  const float* Wr2 = (const float*)d_in[28]; const float* br2 = (const float*)d_in[29];
  const float* Wcl = (const float*)d_in[30]; const float* bcl = (const float*)d_in[31];
  float* out = (float*)d_out_v;

  // output layout (flat, return order)
  float* o_r1  = out;           // 100
  float* o_r2  = out + 100;     // 100
  float* o_cls = out + 200;     // 1000
  float* o_at1 = out + 1200;    // 400000
  float* o_at3 = out + 401200;  // 400000
  float* o_wt  = out + 801200;  // 40000
  float* o_hg3 = out + 841200;  // 9600

  // bump allocator over workspace
  char* wp = (char*)d_ws;
  auto alloc = [&](size_t bytes) -> void* {
    void* p = (void*)wp;
    wp += (bytes + 255) & ~(size_t)255;
    return p;
  };
  // packed-B sizes: nct * ksteps * 512 bf16 elements
  __bf16* Wg1b = (__bf16*)alloc(sizeof(__bf16) * 12 * 2 * 512);  // K=63->KP64, Nc=189
  __bf16* Wg2b = (__bf16*)alloc(sizeof(__bf16) * 12 * 2 * 512);  // K=64->KP64, Nc=192
  __bf16* Wc1b = (__bf16*)alloc(sizeof(__bf16) * 6 * 2 * 512);   // K=64->KP64, Nc=96
  __bf16* Wc2b = (__bf16*)alloc(sizeof(__bf16) * 6 * 3 * 512);   // K=96->KP96, Nc=96
  __bf16* Wc3b = (__bf16*)alloc(sizeof(__bf16) * 6 * 2 * 512);   // K=63->KP64, Nc=96
  float* hg1 = (float*)alloc(sizeof(float) * G * 96);
  float* hg2 = (float*)alloc(sizeof(float) * G * 96);
  float* hg3 = (float*)alloc(sizeof(float) * G * 96);
  float* cnt = (float*)alloc(sizeof(float) * G);
  float*    z    = (float*)alloc(sizeof(float) * (size_t)N * 192);  // padded ldz = 192
  float*    el   = (float*)alloc(sizeof(float) * (size_t)N * 3);
  float*    er   = (float*)alloc(sizeof(float) * (size_t)N * 3);
  unsigned* mseg = (unsigned*)alloc(sizeof(unsigned) * (size_t)N * 3);
  float*    den  = (float*)alloc(sizeof(float) * (size_t)N * 3);
  float*    ebuf = (float*)alloc(sizeof(float) * (size_t)E * 3);
  float*    agg  = (float*)alloc(sizeof(float) * (size_t)N * 192);
  float*    xa   = (float*)alloc(sizeof(float) * (size_t)N * 96);
  float*    xb   = (float*)alloc(sizeof(float) * (size_t)N * 96);
  float*    dou  = (float*)alloc(sizeof(float) * (size_t)N);
  float*    din  = (float*)alloc(sizeof(float) * (size_t)N);

  auto packB = [&](const float* W, __bf16* Bp, int K, int Nc, int KP) {
    int total = ((Nc + 15) >> 4) * (KP >> 5) * 512;
    k_pack_B<<<(total + 255) / 256, 256, 0, stream>>>(W, Bp, K, Nc, KP);
  };
  packB(Wg1, Wg1b, 63, 189, 64);
  packB(Wg2, Wg2b, 64, 192, 64);
  packB(Wc1, Wc1b, 64, 96, 64);
  packB(Wc2, Wc2b, 96, 96, 96);
  packB(Wc3, Wc3b, 63, 96, 64);

  // C gets ldc = Nc rounded up to 16 (padded scratch); grid covers all padded tiles.
  auto gemm = [&](const float* A, const float* rs, const __bf16* Bp, const float* bias,
                  float* C, int M, int K, int Nc, int relu) {
    int nct = (Nc + 15) >> 4;
    int ldc = nct * 16;
    dim3 g(M / 16, (nct + 3) / 4);
    if (K <= 64)
      gemm_bf16_wmma<64><<<g, 128, 0, stream>>>(A, rs, Bp, bias, C, M, K, ldc, relu);
    else
      gemm_bf16_wmma<96><<<g, 128, 0, stream>>>(A, rs, Bp, bias, C, M, K, ldc, relu);
  };

  auto run_graph = [&](const float* h, int Din, const __bf16* Wgb, const float* al,
                       const float* ar, const float* bg, const int* src, const int* dst,
                       const int* gid, int relu_hm, const __bf16* W1b, const float* b1,
                       float* hg, float* attn_out, float* weight_out) {
    const int Dh = Din, HD = HH * Dh;
    const int ldz = (HD + 15) & ~15;  // 192 for both graph types
    // GAT: z = h @ Wg  (no bias pre-aggregation); z padded to ldz cols
    gemm(h, nullptr, Wgb, nullptr, z, N, Din, HD, 0);
    k_el_er<<<(N * HH + 255) / 256, 256, 0, stream>>>(z, al, ar, el, er, N, Dh, ldz);
    k_fill_u32<<<(N * HH + 255) / 256, 256, 0, stream>>>(mseg, 0u, N * HH);
    k_zero_f32<<<(N * HH + 255) / 256, 256, 0, stream>>>(den, N * HH);
    k_edge_max<<<(E * HH + 255) / 256, 256, 0, stream>>>(el, er, src, dst, ebuf, mseg, E);
    k_edge_exp<<<(E * HH + 255) / 256, 256, 0, stream>>>(ebuf, mseg, dst, den, E);
    k_zero_f32<<<(N * HD + 255) / 256, 256, 0, stream>>>(agg, N * HD);
    k_edge_aggr<<<E, 192, 0, stream>>>(ebuf, den, src, dst, z, agg, attn_out, E, Dh, ldz);
    k_bias_headmax<<<(N * Dh + 255) / 256, 256, 0, stream>>>(agg, bg, xa, N, Dh, relu_hm);
    // degrees (shared by both GCN layers)
    k_zero_f32<<<(N + 255) / 256, 256, 0, stream>>>(dou, N);
    k_zero_f32<<<(N + 255) / 256, 256, 0, stream>>>(din, N);
    k_degrees<<<(E + 255) / 256, 256, 0, stream>>>(src, dst, dou, din, E);
    k_rsqrt_clip<<<(N + 255) / 256, 256, 0, stream>>>(dou, N);
    k_rsqrt_clip<<<(N + 255) / 256, 256, 0, stream>>>(din, N);
    // GCN layer 1: Dh -> 96 ; rsqrt(din) folded into GEMM rowScale
    k_zero_f32<<<(N * 96 + 255) / 256, 256, 0, stream>>>(agg, N * 96);
    k_gcn_scatter<<<E, 96, 0, stream>>>(xa, dou, src, dst, agg, E, Dh);
    gemm(agg, din, W1b, b1, xb, N, Dh, 96, 1);
    // GCN layer 2: 96 -> 96 (Wc2)
    k_zero_f32<<<(N * 96 + 255) / 256, 256, 0, stream>>>(agg, N * 96);
    k_gcn_scatter<<<E, 96, 0, stream>>>(xb, dou, src, dst, agg, E, 96);
    gemm(agg, din, Wc2b, bc2, xa, N, 96, 96, 1);
    // mean pool per graph id
    k_zero_f32<<<(G * 96 + 255) / 256, 256, 0, stream>>>(hg, G * 96);
    k_zero_f32<<<1, 128, 0, stream>>>(cnt, G);
    k_pool_sum<<<(N * 96 + 255) / 256, 256, 0, stream>>>(xa, gid, hg, cnt, N, 96);
    k_pool_div<<<(G * 96 + 255) / 256, 256, 0, stream>>>(hg, cnt, G, 96);
    if (weight_out)
      k_copy_col<<<(N + 255) / 256, 256, 0, stream>>>(xa, weight_out, N, 96, 78);
  };

  // graph 1: GAT(Wg1) with relu after head-max; GCN Wc3 then Wc2; attn_values out
  run_graph(h1, 63, Wg1b, al1, ar1, bg1, src1, dst1, gid1, 1, Wc3b, bc3, hg1, o_at1, nullptr);
  // graph 2: same GAT weights, no relu after head-max
  run_graph(h2, 63, Wg1b, al1, ar1, bg1, src2, dst2, gid2, 0, Wc3b, bc3, hg2, nullptr, nullptr);
  // graph 3: GAT(Wg2), GCN Wc1 then Wc2; attn_react + weight column out
  run_graph(h3, 64, Wg2b, al2, ar2, bg2, src3, dst3, gid3, 0, Wc1b, bc1, hg3, o_at3, o_wt);

  // readout heads
  k_readout<<<1, 128, 0, stream>>>(hg1, Wr1, br1, o_r1, G, 96, 1);
  k_readout<<<1, 128, 0, stream>>>(hg2, Wr2, br2, o_r2, G, 96, 1);
  k_readout<<<(G * 10 + 255) / 256, 256, 0, stream>>>(hg3, Wcl, bcl, o_cls, G, 96, 10);
  k_copy<<<(G * 96 + 255) / 256, 256, 0, stream>>>(hg3, o_hg3, G * 96);
}
